// CA_64948495450630
// MI455X (gfx1250) — matrix operand order repaired and checked by reading, not compiled, not run on hardware
//
#include <hip/hip_runtime.h>

// ---------------------------------------------------------------------------
// Neural-CA step for MI455X (gfx1250, wave32, WMMA).
// Per-pixel MLP (48->128 relu ->16) done with v_wmma_f32_16x16x32_f16.
// M-tile = 16 pixels per wave; fragments staged through LDS in the ISA
// register layouts (A: K=(e&7)+16*(e>>3)+8*half per lane-row; B: lane=column,
// K=(e&7)+16*(e>>3)+8*half).  Everything else fp32, coalesced global I/O.
// ---------------------------------------------------------------------------

typedef _Float16 v8h  __attribute__((ext_vector_type(8)));
typedef _Float16 v16h __attribute__((ext_vector_type(16)));
typedef float    v8f  __attribute__((ext_vector_type(8)));

#define NB   4
#define NC   16
#define NH   512
#define NW   512
#define NHID 128
#define HW   (NH * NW)          // 262144 = 1<<18

static __device__ __forceinline__ float ldx0(const float* __restrict__ p, int hh, int ww) {
  return ((unsigned)hh < (unsigned)NH && (unsigned)ww < (unsigned)NW)
             ? p[(hh << 9) + ww] : 0.0f;
}

// Build a v16h fragment from two 16-byte LDS chunks.
static __device__ __forceinline__ v16h frag2(const _Float16* p0, const _Float16* p1) {
  v8h lo = *(const v8h*)p0;
  v8h hi = *(const v8h*)p1;
  return __builtin_shufflevector(lo, hi, 0,1,2,3,4,5,6,7,8,9,10,11,12,13,14,15);
}

static __device__ __forceinline__ v8f wmma16(v16h a, v16h b, v8f c) {
  // (neg_a, A, neg_b, B, c_mod, C, reuse_a, reuse_b)
  return __builtin_amdgcn_wmma_f32_16x16x32_f16(false, a, false, b, (short)0, c,
                                                false, false);
}

__global__ __launch_bounds__(256) void nca_step_kernel(
    const float* __restrict__ x, const float* __restrict__ noise,
    const float* __restrict__ w1, const float* __restrict__ b1,
    const float* __restrict__ w2, const float* __restrict__ b2,
    float* __restrict__ outs, float* __restrict__ outv) {
  // ---- LDS layout (58112 B total) ------------------------------------
  //  [    0, 18432)  W1T  f16[128][72]  (K padded 48->64, row stride 72)
  //                  (overlaid by UPDS f32[8][16][17] after layer-1 sync)
  //  [18432, 22784)  W2T  f16[16][136]
  //  [22784, 57600)  per-wave scratch f16[16][136]:
  //                     phase A/1: PERC f16[16][72]   (K padded 48->64)
  //                     layer 1/2: H1   f16[16][136]  (overlaps PERC, same wave)
  //  [57600, 58112)  ALIVE f32[128]
  __shared__ __align__(16) char SMEM[58112];
  _Float16* W1T  = (_Float16*)SMEM;
  float*    UPDS = (float*)SMEM;                  // overlay, after sync2
  _Float16* W2T  = (_Float16*)(SMEM + 18432);
  _Float16* HPW  = (_Float16*)(SMEM + 22784);
  float*    ALIVE = (float*)(SMEM + 57600);

  const int tid  = threadIdx.x;
  const int wv   = tid >> 5;
  const int l    = tid & 31;
  const int ln   = l & 15;       // lane row/column index inside fragment
  const int half = l >> 4;       // lane-group (0: lanes 0-15, 1: lanes 16-31)
  const int b    = blockIdx.z;
  const int h    = blockIdx.y;
  const int w0   = blockIdx.x * 128 + wv * 16;   // 16 pixels per wave

  // ---- preload weights as f16 into LDS -------------------------------
  for (int i = tid; i < 128 * 64; i += 256) {
    int d = i >> 6, k = i & 63;
    W1T[d * 72 + k] = (k < 48) ? (_Float16)w1[d * 48 + k] : (_Float16)0.0f;
  }
  for (int i = tid; i < 16 * 128; i += 256) {
    int c = i >> 7, d = i & 127;
    W2T[c * 136 + d] = (_Float16)w2[c * 128 + d];
  }
  __syncthreads();

  _Float16* P  = HPW + wv * 2176;   // phase-A perc,  stride 72
  _Float16* Hs = P;                 // h1 restage,    stride 136 (same region)
  float*    U  = UPDS + wv * 272;   // upd staging,   stride 17

  // ---- Phase A: perc (sobel-x / sobel-y / identity) -> f16 LDS --------
  {
    const int wp = w0 + ln;
#pragma unroll
    for (int j = 0; j < 8; ++j) {
      int c = half * 8 + j;                       // halves split the 16 channels
      const float* pc = x + ((size_t)(b * NC + c) << 18);
      float tl = ldx0(pc, h - 1, wp - 1), tm = ldx0(pc, h - 1, wp), tr = ldx0(pc, h - 1, wp + 1);
      float mL = ldx0(pc, h,     wp - 1), mc = ldx0(pc, h,     wp), mR = ldx0(pc, h,     wp + 1);
      float bl = ldx0(pc, h + 1, wp - 1), bm = ldx0(pc, h + 1, wp), br = ldx0(pc, h + 1, wp + 1);
      float gx = ((tr - tl) + 2.0f * (mR - mL) + (br - bl)) * 0.125f;
      float gy = ((bl - tl) + 2.0f * (bm - tm) + (br - tr)) * 0.125f;
      P[ln * 72 +      c] = (_Float16)gx;
      P[ln * 72 + 16 + c] = (_Float16)gy;
      P[ln * 72 + 32 + c] = (_Float16)mc;
      P[ln * 72 + 48 + c] = (_Float16)0.0f;       // zero K-pad 48..63
    }
    if (half == 0) {                              // alive mask from x[0,3] maxpool
      const float* pa = x + ((size_t)3 << 18);
      float mp = -3.0e38f;
#pragma unroll
      for (int dh = -1; dh <= 1; ++dh)
#pragma unroll
        for (int dw = -1; dw <= 1; ++dw) {
          int hh = h + dh, ww = wp + dw;
          if ((unsigned)hh < (unsigned)NH && (unsigned)ww < (unsigned)NW)
            mp = fmaxf(mp, pa[(hh << 9) + ww]);
        }
      ALIVE[wv * 16 + ln] = (mp > 0.1f) ? 1.0f : 0.0f;
    }
  }
  __syncthreads();

  // ---- Layer 1: h1 = relu(perc @ w1^T + b1)  (16x48 . 48x128) ---------
  // A fragments (16-bit A 16x32): lane row = ln, K = (e&7)+16*(e>>3)+8*half.
  v16h a0 = frag2(&P[ln * 72 +      8 * half], &P[ln * 72 +      8 * half + 16]);
  v16h a1 = frag2(&P[ln * 72 + 32 + 8 * half], &P[ln * 72 + 32 + 8 * half + 16]);

#pragma unroll
  for (int nt = 0; nt < 8; ++nt) {
    int d = nt * 16 + ln;                         // output channel = B column
    // B fragments (16-bit B 32x16): lane = column, K = (e&7)+16*(e>>3)+8*half
    v16h bB0 = frag2(&W1T[d * 72 +      8 * half], &W1T[d * 72 +      8 * half + 16]);
    v16h bB1 = frag2(&W1T[d * 72 + 32 + 8 * half], &W1T[d * 72 + 32 + 8 * half + 16]);
    v8f acc = {};
    acc = wmma16(a0, bB0, acc);
    acc = wmma16(a1, bB1, acc);
    float bias = b1[d];
#pragma unroll
    for (int r = 0; r < 8; ++r) {                 // D: lane=N, vgpr r -> M=r+8*half
      float hv = acc[r] + bias;
      hv = hv > 0.0f ? hv : 0.0f;
      Hs[(r + 8 * half) * 136 + d] = (_Float16)hv;
    }
  }
  __syncthreads();   // after this: W1T dead (UPDS may overlay), H1 visible

  // ---- Layer 2: upd = h1 @ w2^T + b2  (16x128 . 128x16) ---------------
  v8f acc2 = {};
#pragma unroll
  for (int q = 0; q < 4; ++q) {
    int kb = q * 32;
    v16h a  = frag2(&Hs[ln * 136 + kb + 8 * half],  &Hs[ln * 136 + kb + 8 * half + 16]);
    v16h bf = frag2(&W2T[ln * 136 + kb + 8 * half], &W2T[ln * 136 + kb + 8 * half + 16]);
    acc2 = wmma16(a, bf, acc2);
  }
  {
    float bias2 = b2[ln];                         // lane = output channel
#pragma unroll
    for (int r = 0; r < 8; ++r)
      U[(r + 8 * half) * 17 + ln] = acc2[r] + bias2;
  }
  __syncthreads();

  // ---- Epilogue: state = (x + upd*(noise>0.5)) * alive; coalesced out --
  {
    const int wpx = w0 + ln;                      // lanes 0-15 contiguous in w
    const float alv = ALIVE[wv * 16 + ln];
#pragma unroll
    for (int i = 0; i < 8; ++i) {
      int c  = i * 2 + half;                      // two channels per iteration
      int gi = (((b * NC + c) << 9) + h) * 512 + wpx;
      float xv = x[gi];
      float nv = noise[gi];
      float uv = U[ln * 17 + c];
      float st = (xv + ((nv > 0.5f) ? uv : 0.0f)) * alv;
      outs[gi] = st;
      if (c < 4)
        outv[(((b * 4 + c) << 9) + h) * 512 + wpx] = st;
    }
  }
}

extern "C" void kernel_launch(void* const* d_in, const int* in_sizes, int n_in,
                              void* d_out, int out_size, void* d_ws, size_t ws_size,
                              hipStream_t stream) {
  const float* x     = (const float*)d_in[0];
  const float* noise = (const float*)d_in[1];
  const float* w1    = (const float*)d_in[2];
  const float* b1    = (const float*)d_in[3];
  const float* w2    = (const float*)d_in[4];
  const float* b2    = (const float*)d_in[5];
  float* outs = (float*)d_out;                       // state: 4*16*512*512
  float* outv = outs + (size_t)NB * NC * NH * NW;    // visible: 4*4*512*512

  dim3 grid(NW / 128, NH, NB);   // 4 x 512 x 4 blocks
  dim3 block(256);               // 8 waves, 16 pixels each
  nca_step_kernel<<<grid, block, 0, stream>>>(x, noise, w1, b1, w2, b2, outs, outv);
}
